// Attention_72318659330705
// MI455X (gfx1250) — matrix operand order repaired and checked by reading, not compiled, not run on hardware
//
#include <hip/hip_runtime.h>

// ---------------------------------------------------------------------------
// Vision attention block for MI455X (gfx1250, wave32, WMMA).
//   B=4, N=1369, DIM=1024, VDIM=1536, H=16, HD=64
// Pipeline (~88 GFLOP, ~150 MB traffic -> bandwidth/latency bound):
//   1) gemm<float,false>  : Q = x @ wq^T                  -> f16 [B,H,N,64]
//   2) gemm<float,true>   : K,V = value @ {wk,wv}^T fused -> f16 [B,H,N,64] x2
//      (shared A tile: value is read once, not twice)
//   3) rope_scale_kernel  : interleaved RoPE on q,k (q also * 64^-0.5)
//   4) flash_attn_kernel  : streaming softmax(QK^T+maskbias)V, double-buffered
//      K/V chunks via GLOBAL_LOAD_ASYNC_TO_LDS_B128 + s_wait_asynccnt,
//      row-sum via WMMA-with-ones, row-max via v_permlane16_b32 (VALU).
//   5) ln_kernel          : LayerNorm -> f16 [B,N,1024]
//   6) gemm<f16,false>    : out = ln @ wp^T + bp -> f32 d_out
// All matrix math via v_wmma_f32_16x16x32_f16 (f32 accumulate).
// ---------------------------------------------------------------------------

#define B_    4
#define N_    1369
#define DIM_  1024
#define VDIM_ 1536
#define H_    16
#define HD_   64
#define ROWS_ (B_ * N_)          // 5476 token rows

typedef _Float16 h16;
typedef __attribute__((ext_vector_type(16))) _Float16 v16h;
typedef __attribute__((ext_vector_type(8)))  float    v8f;

// --- VGPR layout maps per CDNA5 ISA §7.12.2 (16-bit, wave32) ----
__device__ __forceinline__ int a_kmap(int lane, int h) {
  return (h & 7) + ((h >> 3) << 4) + ((lane >> 4) << 3);
}
__device__ __forceinline__ int b_kmap(int lane, int h) {
  return (h & 7) + ((h >> 3) << 4) + ((lane >> 4) << 3);
}
// C/D 16x16 f32: vgpr r, lane l -> M = r + 8*(l>>4), N = l&15

__device__ __forceinline__ v8f vzero8() {
  v8f z = {0.f, 0.f, 0.f, 0.f, 0.f, 0.f, 0.f, 0.f};
  return z;
}

// --- max-reduce across the 16-lane half -----------------------------------
// Butterfly with v_permlane16_b32 (VALU path; selector nibble i = i^off),
// falling back to __shfl_xor (ds_bpermute) if the builtin is unavailable.
__device__ __forceinline__ float rowmax16(float x) {
#if __has_builtin(__builtin_amdgcn_permlane16)
  unsigned v, w;
  v = __float_as_uint(x);
  w = __builtin_amdgcn_permlane16(v, v, 0x67452301u, 0xEFCDAB89u, false, false);
  x = fmaxf(x, __uint_as_float(w)); v = __float_as_uint(x);
  w = __builtin_amdgcn_permlane16(v, v, 0x54761032u, 0xDCFE98BAu, false, false);
  x = fmaxf(x, __uint_as_float(w)); v = __float_as_uint(x);
  w = __builtin_amdgcn_permlane16(v, v, 0x32107654u, 0xBA98FEDCu, false, false);
  x = fmaxf(x, __uint_as_float(w)); v = __float_as_uint(x);
  w = __builtin_amdgcn_permlane16(v, v, 0xFEDCBA98u, 0x76543210u, false, false);
  x = fmaxf(x, __uint_as_float(w));
#else
#pragma unroll
  for (int off = 1; off < 16; off <<= 1)
    x = fmaxf(x, __shfl_xor(x, off, 32));
#endif
  return x;
}

// ---------------------------------------------------------------------------
// GEMM: C[m, o] = sum_k A[m, k] * W[o, k]. Block 64(M)x128(N), K-step 32,
// 256 thr = 8 waves (4x2); wave owns a 16x64 strip (4 C tiles per output).
// DUAL=true computes two outputs sharing the staged A tile (K and V).
// mode 0: f16 head-major [B,H,N,64];  mode 1: f32 row-major + bias.
// ---------------------------------------------------------------------------
template <typename AT, bool DUAL>
__global__ __launch_bounds__(256) void gemm_wmma_nt(
    const AT* __restrict__ A, const float* __restrict__ W0,
    const float* __restrict__ W1, int K, int mode,
    h16* __restrict__ outh0, h16* __restrict__ outh1,
    float* __restrict__ out_f, const float* __restrict__ bias) {
  __shared__ h16 sA[64][32];
  __shared__ h16 sW[DUAL ? 2 : 1][128][32];
  const int tid  = threadIdx.x;
  const int lane = tid & 31;
  const int wid  = tid >> 5;
  const int wm   = wid & 3;
  const int wn   = wid >> 2;
  const int row0 = blockIdx.x * 64;
  const int col0 = blockIdx.y * 128;

  v8f acc[DUAL ? 2 : 1][4];
#pragma unroll
  for (int w = 0; w < (DUAL ? 2 : 1); ++w)
#pragma unroll
    for (int t = 0; t < 4; ++t) acc[w][t] = vzero8();

  for (int k0 = 0; k0 < K; k0 += 32) {
#pragma unroll
    for (int i = tid; i < 64 * 32; i += 256) {
      int r = i >> 5, c = i & 31;
      int gr = row0 + r;
      float v = (gr < ROWS_) ? (float)A[gr * K + k0 + c] : 0.0f;
      sA[r][c] = (h16)v;
    }
#pragma unroll
    for (int i = tid; i < 128 * 32; i += 256) {
      int r = i >> 5, c = i & 31;
      sW[0][r][c] = (h16)W0[(col0 + r) * K + k0 + c];
      if (DUAL) sW[1][r][c] = (h16)W1[(col0 + r) * K + k0 + c];
    }
    __syncthreads();

    v16h af;
    const int am = wm * 16 + (lane & 15);
#pragma unroll
    for (int h = 0; h < 16; ++h) af[h] = sA[am][a_kmap(lane, h)];

#pragma unroll
    for (int t = 0; t < 4; ++t) {
      const int bc = wn * 64 + t * 16 + (lane & 15);
      v16h bf;
#pragma unroll
      for (int h = 0; h < 16; ++h) bf[h] = sW[0][bc][b_kmap(lane, h)];
      acc[0][t] = __builtin_amdgcn_wmma_f32_16x16x32_f16(
          false, af, false, bf, (short)0, acc[0][t], false, false);
      if (DUAL) {
        v16h bf2;
#pragma unroll
        for (int h = 0; h < 16; ++h) bf2[h] = sW[1][bc][b_kmap(lane, h)];
        acc[1][t] = __builtin_amdgcn_wmma_f32_16x16x32_f16(
            false, af, false, bf2, (short)0, acc[1][t], false, false);
      }
    }
    __syncthreads();
  }

#pragma unroll
  for (int w = 0; w < (DUAL ? 2 : 1); ++w) {
    h16* oh = (w == 0) ? outh0 : outh1;
#pragma unroll
    for (int t = 0; t < 4; ++t) {
#pragma unroll
      for (int r = 0; r < 8; ++r) {
        int m = row0 + wm * 16 + r + ((lane >> 4) << 3);
        int o = col0 + wn * 64 + t * 16 + (lane & 15);
        if (m < ROWS_) {
          float v = acc[w][t][r];
          if (mode == 0) {
            int b = m / N_, n = m - b * N_;
            int hh = o >> 6, d = o & 63;
            oh[((b * H_ + hh) * N_ + n) * HD_ + d] = (h16)v;
          } else {
            out_f[m * DIM_ + o] = v + bias[o];
          }
        }
      }
    }
  }
}

// ---------------------------------------------------------------------------
// Interleaved RoPE on q,k (f16 [B,H,N,64]) in place; q also scaled by 64^-0.5.
// ---------------------------------------------------------------------------
__global__ void rope_scale_kernel(h16* __restrict__ q, h16* __restrict__ k) {
  const long total = (long)B_ * H_ * N_ * 32;  // pairs
  long idx = (long)blockIdx.x * blockDim.x + threadIdx.x;
  if (idx >= total) return;
  int  pair = (int)(idx & 31);
  long tok  = idx >> 5;
  int  n    = (int)(tok % N_);
  int  grow = n / 37;
  int  gcol = n - grow * 37;
  int  d0   = pair * 2;
  float t   = (float)((d0 < 32) ? grow : gcol);
  int   j   = pair & 15;
  float inv = __expf(-(float)j * 0.575646273248511f);  // ln(10000)/16
  float s, c;
  __sincosf(t * inv, &s, &c);
  long base = tok * 64 + d0;
  float q0 = (float)q[base], q1 = (float)q[base + 1];
  float k0 = (float)k[base], k1 = (float)k[base + 1];
  const float sc = 0.125f;  // HD^-0.5
  q[base]     = (h16)((q0 * c - q1 * s) * sc);
  q[base + 1] = (h16)((q1 * c + q0 * s) * sc);
  k[base]     = (h16)(k0 * c - k1 * s);
  k[base + 1] = (h16)(k1 * c + k0 * s);
}

// ---------------------------------------------------------------------------
// Async copy of one 32x64 f16 K/V chunk pair into LDS (double buffer slot).
// 512 16-byte units across 128 threads -> 4 GLOBAL_LOAD_ASYNC_TO_LDS_B128
// per wave per chunk (ASYNCcnt += 4/wave). OOB rows clamp (nullified by P=0).
// ---------------------------------------------------------------------------
__device__ __forceinline__ void issue_chunk_async(const h16* __restrict__ Kb,
                                                  const h16* __restrict__ Vb,
                                                  h16 (*dK)[64], h16 (*dV)[64],
                                                  int kc, int tid) {
#pragma unroll
  for (int u = tid; u < 512; u += 128) {
    int which = u >> 8;           // 0 = K, 1 = V
    int idx   = u & 255;          // 16B unit within the 4KB buffer
    int row   = idx >> 3;         // 0..31
    int colh  = (idx & 7) * 8;    // half-element offset 0..56
    int gk    = kc + row;
    if (gk > N_ - 1) gk = N_ - 1;
    const h16* g = (which ? Vb : Kb) + gk * 64 + colh;
    h16*       l = (which ? dV[row] : dK[row]) + colh;
    unsigned lds          = (unsigned)(unsigned long long)(void*)l;
    unsigned long long ga = (unsigned long long)(const void*)g;
    asm volatile("global_load_async_to_lds_b128 %0, %1, off"
                 :: "v"(lds), "v"(ga) : "memory");
  }
}

// ---------------------------------------------------------------------------
// Flash attention: blockIdx.y = b*16+h, 64 query rows per blockIdx.x,
// 128 thr = 4 waves x 16 query rows. K/V streamed in 32-key chunks, double
// buffered with async-to-LDS. Per chunk/wave: 4 WMMA (S) + 1 WMMA (rowsum)
// + 4 WMMA (PV). Online softmax; row max via v_permlane16_b32.
// ---------------------------------------------------------------------------
__global__ __launch_bounds__(128) void flash_attn_kernel(
    const h16* __restrict__ qh, const h16* __restrict__ kh,
    const h16* __restrict__ vh, const unsigned char* __restrict__ mask,
    float* __restrict__ outp /* [B, N, 1024] */) {
  __shared__ h16 sK[2][32][64];
  __shared__ h16 sV[2][32][64];
  __shared__ h16 sP[4][16][32];      // per-wave P restriping patch

  const int tid = threadIdx.x, lane = tid & 31, wid = tid >> 5;
  const int bh  = blockIdx.y;
  const int b   = bh >> 4;
  const int hh  = bh & 15;
  const int q0  = blockIdx.x * 64 + wid * 16;

  const h16* Qb = qh + (long)bh * N_ * 64;
  const h16* Kb = kh + (long)bh * N_ * 64;
  const h16* Vb = vh + (long)bh * N_ * 64;
  const unsigned char* maskp = mask + b * N_;

  // Q fragments (K-dim 64 -> two 16x32 A frags), loaded once.
  v16h qa[2];
  {
    int qm  = q0 + (lane & 15);
    int qmc = (qm < N_) ? qm : (N_ - 1);
#pragma unroll
    for (int f = 0; f < 2; ++f)
#pragma unroll
      for (int h = 0; h < 16; ++h)
        qa[f][h] = Qb[qmc * 64 + f * 32 + a_kmap(lane, h)];
  }
  v16h vones;
#pragma unroll
  for (int h = 0; h < 16; ++h) vones[h] = (h16)1.0f;

  float rm[8], rl[8], alpha[8];
#pragma unroll
  for (int r = 0; r < 8; ++r) { rm[r] = -1e30f; rl[r] = 0.0f; }
  v8f o[4];
#pragma unroll
  for (int t = 0; t < 4; ++t) o[t] = vzero8();

  issue_chunk_async(Kb, Vb, sK[0], sV[0], 0, tid);  // prologue fill

  const int nchunks = (N_ + 31) / 32;               // 43
  for (int c = 0; c < nchunks; ++c) {
    const int kc  = c * 32;
    const int cur = c & 1;
    const bool have_next = (c + 1) < nchunks;
    if (have_next)
      issue_chunk_async(Kb, Vb, sK[1 - cur], sV[1 - cur], kc + 32, tid);
    if (have_next)
      asm volatile("s_wait_asynccnt 0x4" ::: "memory");  // chunk c landed
    else
      asm volatile("s_wait_asynccnt 0x0" ::: "memory");
    __syncthreads();  // all waves' chunk-c async fills visible

    // per-lane mask bias for this chunk's two key columns
    int key0 = kc + (lane & 15), key1 = key0 + 16;
    float mb0 = (key0 < N_ && maskp[(key0 < N_) ? key0 : 0]) ? 0.f : -1e30f;
    float mb1 = (key1 < N_ && maskp[(key1 < N_) ? key1 : 0]) ? 0.f : -1e30f;

    // S = Q K^T : two 16x16 tiles, K-dim 64 = 2 WMMA steps each
    v8f s0 = vzero8(), s1 = vzero8();
#pragma unroll
    for (int f = 0; f < 2; ++f) {
      v16h bf0, bf1;
      int key = lane & 15;
#pragma unroll
      for (int h = 0; h < 16; ++h) {
        bf0[h] = sK[cur][key][f * 32 + b_kmap(lane, h)];
        bf1[h] = sK[cur][16 + key][f * 32 + b_kmap(lane, h)];
      }
      s0 = __builtin_amdgcn_wmma_f32_16x16x32_f16(false, qa[f], false, bf0,
                                                  (short)0, s0, false, false);
      s1 = __builtin_amdgcn_wmma_f32_16x16x32_f16(false, qa[f], false, bf1,
                                                  (short)0, s1, false, false);
    }

    // online softmax: row = r + 8*(lane>>4); stats uniform across the half
#pragma unroll
    for (int r = 0; r < 8; ++r) {
      float v0 = s0[r] + mb0, v1 = s1[r] + mb1;
      float mx = rowmax16(fmaxf(v0, v1));
      float mnew = fmaxf(rm[r], mx);
      alpha[r] = __expf(rm[r] - mnew);
      rm[r] = mnew;
      s0[r] = __expf(v0 - mnew);
      s1[r] = __expf(v1 - mnew);
#pragma unroll
      for (int t = 0; t < 4; ++t) o[t][r] *= alpha[r];
    }

    // restripe P (C-layout) -> A-layout through wave-private LDS
#pragma unroll
    for (int r = 0; r < 8; ++r) {
      int m = r + ((lane >> 4) << 3);
      sP[wid][m][lane & 15]        = (h16)s0[r];
      sP[wid][m][16 + (lane & 15)] = (h16)s1[r];
    }
    asm volatile("s_wait_dscnt 0x0" ::: "memory");  // wave-local LDS RAW fence
    v16h pf;
#pragma unroll
    for (int h = 0; h < 16; ++h)
      pf[h] = sP[wid][lane & 15][a_kmap(lane, h)];

    // row sums via WMMA: P(16x32) @ ones(32x16) -> every column = rowsum
    v8f rs = __builtin_amdgcn_wmma_f32_16x16x32_f16(false, pf, false, vones,
                                                    (short)0, vzero8(), false,
                                                    false);
#pragma unroll
    for (int r = 0; r < 8; ++r) rl[r] = rl[r] * alpha[r] + rs[r];

    // O += P(16x32) @ V(32x64)
#pragma unroll
    for (int t = 0; t < 4; ++t) {
      v16h vf;
#pragma unroll
      for (int h = 0; h < 16; ++h)
        vf[h] = sV[cur][b_kmap(lane, h)][t * 16 + (lane & 15)];
      o[t] = __builtin_amdgcn_wmma_f32_16x16x32_f16(false, pf, false, vf,
                                                    (short)0, o[t], false,
                                                    false);
    }
    __syncthreads();  // buffer `cur` free before chunk c+2 refills it
  }

  // normalize + store token-major [B, N, H*HD]
#pragma unroll
  for (int t = 0; t < 4; ++t) {
#pragma unroll
    for (int r = 0; r < 8; ++r) {
      int m = q0 + r + ((lane >> 4) << 3);
      if (m < N_) {
        int d = t * 16 + (lane & 15);
        float val = o[t][r] / fmaxf(rl[r], 1e-20f);
        outp[(b * N_ + m) * DIM_ + hh * HD_ + d] = val;
      }
    }
  }
}

// ---------------------------------------------------------------------------
// LayerNorm over last dim (1024), one 256-thread block per token; f16 out.
// ---------------------------------------------------------------------------
__global__ __launch_bounds__(256) void ln_kernel(
    const float* __restrict__ in, const float* __restrict__ g,
    const float* __restrict__ bta, h16* __restrict__ out) {
  const int row = blockIdx.x;
  const float* p = in + (long)row * DIM_;
  float s = 0.f, s2 = 0.f;
  for (int i = threadIdx.x; i < DIM_; i += 256) {
    float v = p[i];
    s += v; s2 += v * v;
  }
  __shared__ float red[2][8];
#pragma unroll
  for (int off = 16; off; off >>= 1) {
    s  += __shfl_xor(s, off, 32);
    s2 += __shfl_xor(s2, off, 32);
  }
  int lane = threadIdx.x & 31, wid = threadIdx.x >> 5;
  if (lane == 0) { red[0][wid] = s; red[1][wid] = s2; }
  __syncthreads();
  if (wid == 0) {
    s  = (lane < 8) ? red[0][lane] : 0.f;
    s2 = (lane < 8) ? red[1][lane] : 0.f;
#pragma unroll
    for (int off = 4; off; off >>= 1) {
      s  += __shfl_xor(s, off, 32);
      s2 += __shfl_xor(s2, off, 32);
    }
    if (lane == 0) { red[0][0] = s; red[1][0] = s2; }
  }
  __syncthreads();
  float mean = red[0][0] * (1.0f / DIM_);
  float var  = red[1][0] * (1.0f / DIM_) - mean * mean;
  float inv  = rsqrtf(var + 1e-5f);
  h16* q = out + (long)row * DIM_;
  for (int i = threadIdx.x; i < DIM_; i += 256)
    q[i] = (h16)((p[i] - mean) * inv * g[i] + bta[i]);
}

// ---------------------------------------------------------------------------
extern "C" void kernel_launch(void* const* d_in, const int* in_sizes, int n_in,
                              void* d_out, int out_size, void* d_ws,
                              size_t ws_size, hipStream_t stream) {
  const float* x     = (const float*)d_in[0];
  const float* value = (const float*)d_in[1];
  const unsigned char* mask = (const unsigned char*)d_in[2];  // jax bool = 1B
  const float* wq   = (const float*)d_in[3];
  const float* wk   = (const float*)d_in[4];
  const float* wv   = (const float*)d_in[5];
  const float* ln_g = (const float*)d_in[6];
  const float* ln_b = (const float*)d_in[7];
  const float* wp   = (const float*)d_in[8];
  const float* bp   = (const float*)d_in[9];
  float* out = (float*)d_out;

  // workspace carve-up (~64 MB, every byte rewritten each call)
  const size_t qkv_elems = (size_t)B_ * H_ * N_ * HD_;  // 5,607,424
  char* ws = (char*)d_ws;
  h16* qh = (h16*)ws;            ws += qkv_elems * sizeof(h16);
  h16* kh = (h16*)ws;            ws += qkv_elems * sizeof(h16);
  h16* vh = (h16*)ws;            ws += qkv_elems * sizeof(h16);
  float* attn_out = (float*)ws;  ws += (size_t)ROWS_ * DIM_ * sizeof(float);
  h16* ln_out = (h16*)ws;

  dim3 gblk(256);
  dim3 ggrid((ROWS_ + 63) / 64, DIM_ / 128);  // 86 x 8

  gemm_wmma_nt<float, false><<<ggrid, gblk, 0, stream>>>(
      x, wq, nullptr, DIM_, 0, qh, nullptr, nullptr, nullptr);
  gemm_wmma_nt<float, true><<<ggrid, gblk, 0, stream>>>(
      value, wk, wv, VDIM_, 0, kh, vh, nullptr, nullptr);

  long pairs = (long)B_ * H_ * N_ * 32;
  rope_scale_kernel<<<(unsigned)((pairs + 255) / 256), 256, 0, stream>>>(qh, kh);

  flash_attn_kernel<<<dim3((N_ + 63) / 64, B_ * H_), 128, 0, stream>>>(
      qh, kh, vh, mask, attn_out);

  ln_kernel<<<ROWS_, 256, 0, stream>>>(attn_out, ln_g, ln_b, ln_out);

  gemm_wmma_nt<h16, false><<<ggrid, gblk, 0, stream>>>(
      ln_out, wp, nullptr, DIM_, 1, nullptr, nullptr, out, bp);
}
